// NTXentLoss_35098472743288
// MI455X (gfx1250) — compile-verified, matrix operand order checked by reading, not run on hardware
//
#include <hip/hip_runtime.h>

typedef __attribute__((ext_vector_type(16))) _Float16 v16h;
typedef __attribute__((ext_vector_type(8)))  _Float16 h8;
typedef __attribute__((ext_vector_type(4)))  _Float16 h4;
typedef __attribute__((ext_vector_type(8)))  float    v8f;
typedef __attribute__((ext_vector_type(4)))  unsigned int v4u;
typedef __attribute__((ext_vector_type(4)))  int      v4i;
typedef __attribute__((ext_vector_type(8)))  int      v8i;

#define BROWS 4096
#define DIM   512
#define NTOT  8192
#define TAUF  0.07f

// B tile staged in LDS: 32 rows x 512 halves, padded to 520 halves/row
// (1040B stride -> ds_load_b128 lanes land on rotating bank quads)
#define BPAD        520
#define BTILE_ROWS  32
#define BTILE_BYTES (BTILE_ROWS * BPAD * 2)   /* 33280 */
#define SIM_LDS     (2 * BTILE_BYTES)         /* double buffer: 66560 B */

#ifndef __has_builtin
#define __has_builtin(x) 0
#endif
#if __has_builtin(__builtin_amdgcn_tensor_load_to_lds)
#define HAVE_TDM 1
#else
#define HAVE_TDM 0
#endif
#if defined(__has_include)
#if __has_include(<hip/amd_detail/amd_gfx1250_TDM.h>)
#define TDM_6ARG 1
#else
#define TDM_6ARG 0
#endif
#else
#define TDM_6ARG 0
#endif

union ABFrag { v16h v; h8 h[2]; };

__device__ __forceinline__ void loadFrag(ABFrag& f, const _Float16* row,
                                         int kt, int g) {
  f.h[0] = *(const h8*)(row + kt + g * 8);
  f.h[1] = *(const h8*)(row + kt + 16 + g * 8);
}

#if HAVE_TDM
// TDM: DMA a 32-row x 512-half tile of zsn into LDS at ldsByteOff, inserting
// 4 DWORDs of pad every 256 DWORDs (-> 520-half LDS row stride).
__device__ __forceinline__ void tdm_load_tile(const _Float16* gsrc,
                                              unsigned ldsByteOff) {
  unsigned long long ga = (unsigned long long)gsrc;
  // D# group 0: count=1 | lds_addr | global_addr[56:0] | type=2
  v4u g0 = { 1u,
             ldsByteOff,
             (unsigned)(ga & 0xFFFFFFFFu),
             (unsigned)((ga >> 32) & 0x01FFFFFFu) | (2u << 30) };
  // D# group 1: data_size=2B, pad_enable, pad_interval=256DW, pad_amount=4DW,
  //             tensor_dim0=512, tensor_dim1=8192, tile_dim0=512,
  //             tile_dim1=32, tensor_dim0_stride=512
  v8i g1 = { (int)((1u << 16) | (1u << 20) | (7u << 22) | (3u << 25)),
             (int)(512u << 16),   /* tensor_dim0[15:0] @ bits63:48 */
             (int)(8192u << 16),  /* tensor_dim1[15:0] @ bits95:80 */
             (int)(512u << 16),   /* tile_dim0 @ bits127:112 */
             BTILE_ROWS,          /* tile_dim1 @ bits143:128 */
             512,                 /* tensor_dim0_stride @ bits207:160 */
             0, 0 };
  v4i g2 = { 0, 0, 0, 0 };
  v4i g3 = { 0, 0, 0, 0 };
#if TDM_6ARG
  v8i g4 = { 0, 0, 0, 0, 0, 0, 0, 0 };
  __builtin_amdgcn_tensor_load_to_lds(g0, g1, g2, g3, g4, 0);
#else
  __builtin_amdgcn_tensor_load_to_lds(g0, g1, g2, g3, 0);
#endif
}
#endif

// ---------------- 1) row normalize: z -> f16 zsn, store 1/max(||z||,eps) ----
__global__ __launch_bounds__(256) void nrm_k(const float* __restrict__ z0,
                                             const float* __restrict__ z1,
                                             _Float16* __restrict__ zsn,
                                             float* __restrict__ rs) {
  const int wave = threadIdx.x >> 5, lane = threadIdx.x & 31;
  const int row  = blockIdx.x * 8 + wave;
  const float* src = (row < BROWS) ? (z0 + (size_t)row * DIM)
                                   : (z1 + (size_t)(row - BROWS) * DIM);
  float4 x[4];
  float ss = 0.f;
  const float4* s4 = (const float4*)src;
  #pragma unroll
  for (int j = 0; j < 4; ++j) {
    x[j] = s4[lane + 32 * j];
    ss += x[j].x * x[j].x + x[j].y * x[j].y + x[j].z * x[j].z + x[j].w * x[j].w;
  }
  #pragma unroll
  for (int m = 16; m >= 1; m >>= 1) ss += __shfl_xor(ss, m, 32);
  const float s = 1.0f / fmaxf(sqrtf(ss), 1e-12f);
  _Float16* dst = zsn + (size_t)row * DIM;
  #pragma unroll
  for (int j = 0; j < 4; ++j) {
    h4 o;
    o[0] = (_Float16)(x[j].x * s); o[1] = (_Float16)(x[j].y * s);
    o[2] = (_Float16)(x[j].z * s); o[3] = (_Float16)(x[j].w * s);
    ((h4*)dst)[lane + 32 * j] = o;
  }
  if (lane == 0) rs[row] = s;
}

// ---------------- 2) zero per-row sum-exp accumulators ----------------------
__global__ void zero_k(float* __restrict__ rowsum) {
  int i = blockIdx.x * blockDim.x + threadIdx.x;
  if (i < NTOT) rowsum[i] = 0.0f;
}

// ---------------- 3) f32 positive-pair dots --------------------------------
__global__ __launch_bounds__(256) void pos_k(const float* __restrict__ z0,
                                             const float* __restrict__ z1,
                                             const float* __restrict__ rs,
                                             float* __restrict__ pos) {
  const int wave = threadIdx.x >> 5, lane = threadIdx.x & 31;
  const int i = blockIdx.x * 8 + wave;
  const float4* a = (const float4*)(z0 + (size_t)i * DIM);
  const float4* b = (const float4*)(z1 + (size_t)i * DIM);
  float d = 0.f;
  #pragma unroll
  for (int j = 0; j < 4; ++j) {
    float4 av = a[lane + 32 * j], bv = b[lane + 32 * j];
    d += av.x * bv.x + av.y * bv.y + av.z * bv.z + av.w * bv.w;
  }
  #pragma unroll
  for (int m = 16; m >= 1; m >>= 1) d += __shfl_xor(d, m, 32);
  if (lane == 0) pos[i] = d * rs[i] * rs[i + BROWS];
}

// ---------------- 4) fused WMMA sim tiles + sum exp(sim/tau), diag masked ---
// grid (64, 8), block 256 (8 waves). Each wave owns 16 rows (A K-fragments in
// 128 VGPRs) and two independent 16-col accumulator chains per staged tile.
// All waves share a TDM-staged, double-buffered 32-col B tile in LDS; the
// next tile's DMA overlaps the current 32 WMMAs. B fragments are register
// double-buffered so ds waits are partial and overlap the WMMA stream.
__global__ __launch_bounds__(256) void simlse_k(const _Float16* __restrict__ zsn,
                                                float* __restrict__ rowsum) {
  extern __shared__ char smem[];
  const int tid  = threadIdx.x;
  const int lane = tid & 31, wave = tid >> 5;
  const int g    = lane >> 4;              // half-wave group (K-chunk select)
  const int l15  = lane & 15;
  const int waveRow = blockIdx.x * 128 + wave * 16;
  const int col0    = blockIdx.y * 1024;

  // Preload this wave's 16x512 A panel into registers (16 K-fragments).
  const _Float16* arow = zsn + (size_t)(waveRow + l15) * DIM;
  ABFrag afr[16];
  #pragma unroll
  for (int kk = 0; kk < 16; ++kk) loadFrag(afr[kk], arow, kk * 32, g);

  float facc[8] = {0.f, 0.f, 0.f, 0.f, 0.f, 0.f, 0.f, 0.f};
  const float invTau = 1.0f / TAUF;

#if HAVE_TDM
  if (wave == 0) tdm_load_tile(zsn + (size_t)col0 * DIM, 0u);
#endif

  int buf = 0;
  for (int cs = 0; cs < 1024; cs += BTILE_ROWS, buf ^= 1) {
#if HAVE_TDM
    if (wave == 0) {
      if (cs + BTILE_ROWS < 1024) {
        tdm_load_tile(zsn + (size_t)(col0 + cs + BTILE_ROWS) * DIM,
                      (unsigned)((buf ^ 1) * BTILE_BYTES));
        __builtin_amdgcn_s_wait_tensorcnt(1);   // current buf's DMA done
      } else {
        __builtin_amdgcn_s_wait_tensorcnt(0);
      }
    }
    __syncthreads();                            // publish buf to all waves
#else
    {
      const uint4* src = (const uint4*)(zsn + (size_t)(col0 + cs) * DIM);
      uint4* dst = (uint4*)(smem + buf * BTILE_BYTES);
      #pragma unroll
      for (int it = 0; it < 8; ++it) {
        int idx = tid + it * 256;               // 2048 uint4 = 32 rows x 64
        int r = idx >> 6, cc = idx & 63;
        dst[r * 65 + cc] = src[(size_t)r * 64 + cc];
      }
      __syncthreads();
    }
#endif
    const _Float16* btile = (const _Float16*)(smem + buf * BTILE_BYTES);
    const _Float16* brow0 = btile + (size_t)l15 * BPAD;          // cols +0..15
    const _Float16* brow1 = btile + (size_t)(l15 + 16) * BPAD;   // cols +16..31

    v8f c0 = {}, c1 = {};
    ABFrag b0c, b1c, b0n, b1n;
    loadFrag(b0c, brow0, 0, g);
    loadFrag(b1c, brow1, 0, g);
    #pragma unroll
    for (int kk = 0; kk < 16; ++kk) {           // 32x v_wmma_f32_16x16x32_f16
      if (kk < 15) {                            // prefetch next K-step's B
        loadFrag(b0n, brow0, (kk + 1) * 32, g);
        loadFrag(b1n, brow1, (kk + 1) * 32, g);
      }
      c0 = __builtin_amdgcn_wmma_f32_16x16x32_f16(
             false, afr[kk].v, false, b0c.v, (short)0, c0, false, false);
      c1 = __builtin_amdgcn_wmma_f32_16x16x32_f16(
             false, afr[kk].v, false, b1c.v, (short)0, c1, false, false);
      b0c = b0n;                                // unrolled -> SSA renamed
      b1c = b1n;
    }

    // accumulate exp(sim/tau), skipping the self-similarity diagonal
    const int colA = col0 + cs + l15;
    const int colB = colA + 16;
    #pragma unroll
    for (int k = 0; k < 8; ++k) {
      const int r = waveRow + k + g * 8;        // C layout: VGPR k -> M=k+8g
      const float e0 = __expf(c0[k] * invTau);
      const float e1 = __expf(c1[k] * invTau);
      facc[k] += (r == colA) ? 0.0f : e0;
      facc[k] += (r == colB) ? 0.0f : e1;
    }
    __syncthreads();   // all waves done with buf before it is re-filled
  }

  // reduce each row's partial over the 16 lanes sharing that row
  #pragma unroll
  for (int k = 0; k < 8; ++k) {
    float v = facc[k];
    #pragma unroll
    for (int m = 8; m >= 1; m >>= 1) v += __shfl_xor(v, m, 16);
    if (l15 == 0) atomicAdd(&rowsum[waveRow + k + g * 8], v);
  }
}

// ---------------- 5) final scalar reduction --------------------------------
__global__ __launch_bounds__(256) void fin_k(const float* __restrict__ rowsum,
                                             const float* __restrict__ pos,
                                             float* __restrict__ out) {
  __shared__ float sh[8];
  const int tid = threadIdx.x, lane = tid & 31, wave = tid >> 5;
  float t = 0.f;
  for (int i = tid; i < NTOT; i += 256) t += logf(rowsum[i]);
  float p = 0.f;
  for (int i = tid; i < BROWS; i += 256) p += pos[i];
  t -= 2.0f * p * (1.0f / TAUF);           // each pos appears in two rows
  #pragma unroll
  for (int m = 16; m >= 1; m >>= 1) t += __shfl_xor(t, m, 32);
  if (lane == 0) sh[wave] = t;
  __syncthreads();
  if (tid == 0) {
    float s = 0.f;
    #pragma unroll
    for (int w = 0; w < 8; ++w) s += sh[w];
    out[0] = s / (float)NTOT;
  }
}

extern "C" void kernel_launch(void* const* d_in, const int* in_sizes, int n_in,
                              void* d_out, int out_size, void* d_ws, size_t ws_size,
                              hipStream_t stream) {
  (void)in_sizes; (void)n_in; (void)out_size; (void)ws_size;
  const float* z0 = (const float*)d_in[0];
  const float* z1 = (const float*)d_in[1];
  char* ws = (char*)d_ws;

  _Float16* zsn  = (_Float16*)ws;                              // 8 MB
  float* rs      = (float*)(ws + 8388608);                     // 32 KB
  float* rowsum  = (float*)(ws + 8388608 + 32768);             // 32 KB
  float* pos     = (float*)(ws + 8388608 + 65536);             // 16 KB
  float* out     = (float*)d_out;

  nrm_k   <<<dim3(1024),  256, 0,       stream>>>(z0, z1, zsn, rs);
  zero_k  <<<dim3(32),    256, 0,       stream>>>(rowsum);
  pos_k   <<<dim3(512),   256, 0,       stream>>>(z0, z1, rs, pos);
  simlse_k<<<dim3(64, 8), 256, SIM_LDS, stream>>>(zsn, rowsum);
  fin_k   <<<dim3(1),     256, 0,       stream>>>(rowsum, pos, out);
}
